// MultiHeadAttention_66425964200321
// MI455X (gfx1250) — matrix operand order repaired and checked by reading, not compiled, not run on hardware
//
#include <hip/hip_runtime.h>
#include <hip/hip_bf16.h>
#include <hip/hip_fp16.h>

// ---------------------------------------------------------------------------
// MI455X (gfx1250) causal multi-head attention, wave32 + WMMA f16 pipeline.
//   B=4, S=2048, D=1024, H=16, HD=64
// Stage 1/3: double-buffered LDS GEMM, 128x128 WG tile, 32x64 per wave.
// Stage 2: flash attention, 128 q-rows per WG, 32-key blocks.
// ---------------------------------------------------------------------------

typedef _Float16 h4   __attribute__((ext_vector_type(4)));
typedef _Float16 h8   __attribute__((ext_vector_type(8)));
typedef _Float16 v16h __attribute__((ext_vector_type(16)));
typedef float    f4   __attribute__((ext_vector_type(4)));
typedef float    v8f  __attribute__((ext_vector_type(8)));

static constexpr int Bsz   = 4;
static constexpr int Sseq  = 2048;
static constexpr int Dmod  = 1024;
static constexpr int Hn    = 16;
static constexpr int HDim  = 64;
static constexpr int Mrows = Bsz * Sseq;          // 8192
static constexpr float SCALE = 0.125f;            // 1/sqrt(64)

union V16 { v16h v; h8 h[2]; };

__device__ __forceinline__ v8f wmma_f16(v16h a, v16h b, v8f c) {
  // D = A(16x32 f16) * B(32x16 f16) + C(16x16 f32)
  return __builtin_amdgcn_wmma_f32_16x16x32_f16(false, a, false, b, (short)0, c,
                                                false, false);
}

// ---------------------------------------------------------------------------
// Stage 0: fp32 -> fp16 convert (vectorized x4)
// ---------------------------------------------------------------------------
__global__ void cvt_f32_f16_v4(const f4* __restrict__ in, h4* __restrict__ out,
                               int n4) {
  int i = blockIdx.x * blockDim.x + threadIdx.x;
  if (i < n4) {
    f4 v = in[i];
    h4 o;
    o[0] = (_Float16)v[0]; o[1] = (_Float16)v[1];
    o[2] = (_Float16)v[2]; o[3] = (_Float16)v[3];
    out[i] = o;
  }
}

// ---------------------------------------------------------------------------
// Shared GEMM core: C(128x128 tile) = A16 . B16^T, K=1024, double-buffered
// LDS, register-staged global loads, 8 WMMAs per wave per K-step.
// Each wave: rows rt*32..+32 (2 A frags), cols ch*64..+64 (4 B frags).
// Accumulators returned through acc[2][4].
// ---------------------------------------------------------------------------
#define GEMM_CORE(A_PTR, B_PTR, ACC)                                           \
  __shared__ _Float16 Al[2][128][40];                                          \
  __shared__ _Float16 Bl[2][128][40];                                          \
  const int tid   = threadIdx.x;                                               \
  const int w     = tid >> 5;                                                  \
  const int lane  = tid & 31;                                                  \
  const int l16   = lane & 15;                                                 \
  const int lhalf = (lane < 16) ? 0 : 1;                                       \
  const int rt    = w >> 1;                                                    \
  const int ch    = w & 1;                                                     \
  const int lrow  = tid >> 1;                                                  \
  const int ldg   = tid & 1;                                                   \
  constexpr int Kd = 1024;                                                     \
  v8f ACC[2][4] = {};                                                          \
  const _Float16* aload = A_PTR + (size_t)(mbase + lrow) * Kd + ldg * 16;      \
  const _Float16* bload = B_PTR + (size_t)(nbase + lrow) * Kd + ldg * 16;      \
  {  /* stage K-tile 0 */                                                      \
    const h8* as = (const h8*)aload;                                           \
    const h8* bs = (const h8*)bload;                                           \
    h8 a0 = as[0], a1 = as[1], b0 = bs[0], b1 = bs[1];                         \
    *(h8*)&Al[0][lrow][ldg * 16]     = a0;                                     \
    *(h8*)&Al[0][lrow][ldg * 16 + 8] = a1;                                     \
    *(h8*)&Bl[0][lrow][ldg * 16]     = b0;                                     \
    *(h8*)&Bl[0][lrow][ldg * 16 + 8] = b1;                                     \
  }                                                                            \
  __syncthreads();                                                             \
  int buf = 0;                                                                 \
  for (int kb = 0; kb < Kd - 32; kb += 32) {                                   \
    /* issue next tile's global loads early (overlap with WMMAs) */            \
    const h8* as = (const h8*)(aload + kb + 32);                               \
    const h8* bs = (const h8*)(bload + kb + 32);                               \
    h8 na0 = as[0], na1 = as[1], nb0 = bs[0], nb1 = bs[1];                     \
    __builtin_prefetch(aload + kb + 64, 0, 3);                                 \
    __builtin_prefetch(bload + kb + 64, 0, 3);                                 \
    /* gather all fragments, then back-to-back WMMAs */                        \
    V16 af[2], bf[4];                                                          \
    _Pragma("unroll") for (int r = 0; r < 2; ++r) {                            \
      const _Float16* ap = &Al[buf][rt * 32 + r * 16 + l16][lhalf * 8];        \
      af[r].h[0] = *(const h8*)ap;                                             \
      af[r].h[1] = *(const h8*)(ap + 16);                                      \
    }                                                                          \
    _Pragma("unroll") for (int nt = 0; nt < 4; ++nt) {                         \
      const _Float16* bp = &Bl[buf][ch * 64 + nt * 16 + l16][lhalf * 8];       \
      bf[nt].h[0] = *(const h8*)bp;                                            \
      bf[nt].h[1] = *(const h8*)(bp + 16);                                     \
    }                                                                          \
    _Pragma("unroll") for (int r = 0; r < 2; ++r)                              \
      _Pragma("unroll") for (int nt = 0; nt < 4; ++nt)                         \
        ACC[r][nt] = wmma_f16(af[r].v, bf[nt].v, ACC[r][nt]);                  \
    /* commit next tile to the other LDS buffer */                             \
    *(h8*)&Al[buf ^ 1][lrow][ldg * 16]     = na0;                              \
    *(h8*)&Al[buf ^ 1][lrow][ldg * 16 + 8] = na1;                              \
    *(h8*)&Bl[buf ^ 1][lrow][ldg * 16]     = nb0;                              \
    *(h8*)&Bl[buf ^ 1][lrow][ldg * 16 + 8] = nb1;                              \
    __syncthreads();                                                           \
    buf ^= 1;                                                                  \
  }                                                                            \
  { /* last K-tile */                                                          \
    V16 af[2], bf[4];                                                          \
    _Pragma("unroll") for (int r = 0; r < 2; ++r) {                            \
      const _Float16* ap = &Al[buf][rt * 32 + r * 16 + l16][lhalf * 8];        \
      af[r].h[0] = *(const h8*)ap;                                             \
      af[r].h[1] = *(const h8*)(ap + 16);                                      \
    }                                                                          \
    _Pragma("unroll") for (int nt = 0; nt < 4; ++nt) {                         \
      const _Float16* bp = &Bl[buf][ch * 64 + nt * 16 + l16][lhalf * 8];       \
      bf[nt].h[0] = *(const h8*)bp;                                            \
      bf[nt].h[1] = *(const h8*)(bp + 16);                                     \
    }                                                                          \
    _Pragma("unroll") for (int r = 0; r < 2; ++r)                              \
      _Pragma("unroll") for (int nt = 0; nt < 4; ++nt)                         \
        ACC[r][nt] = wmma_f16(af[r].v, bf[nt].v, ACC[r][nt]);                  \
  }

// ---------------------------------------------------------------------------
// Stage 1: QKV projection GEMM.  M=8192, N=3072, K=1024.
// Epilogue scatters into per-head Q/K/V f16 buffers [B,H,S,HD].
// ---------------------------------------------------------------------------
__global__ __launch_bounds__(256) void gemm_qkv(
    const _Float16* __restrict__ A,    // [8192,1024]
    const _Float16* __restrict__ W,    // [3072,1024]
    const float*    __restrict__ bias, // [3072]
    _Float16* __restrict__ Qb, _Float16* __restrict__ Kb,
    _Float16* __restrict__ Vb) {
  const int mbase = blockIdx.x * 128;
  const int nbase = blockIdx.y * 128;
  GEMM_CORE(A, W, acc)

#pragma unroll
  for (int nt = 0; nt < 4; ++nt) {
    const int n  = nbase + ch * 64 + nt * 16 + l16;
    const float bv = bias[n];
    const int which = n >> 10;
    const int h = (n >> 6) & 15;
    const int d = n & 63;
    _Float16* dst = (which == 0) ? Qb : (which == 1) ? Kb : Vb;
#pragma unroll
    for (int r = 0; r < 2; ++r)
#pragma unroll
      for (int j = 0; j < 8; ++j) {
        const int m = mbase + rt * 32 + r * 16 + j + lhalf * 8;
        const int b = m >> 11;        // / 2048
        const int s = m & 2047;
        dst[(((size_t)(b * Hn + h) * Sseq) + s) * HDim + d] =
            (_Float16)(acc[r][nt][j] + bv);
      }
  }
}

// ---------------------------------------------------------------------------
// Stage 3: output projection GEMM. M=8192, N=1024, K=1024 -> fp32 d_out.
// ---------------------------------------------------------------------------
__global__ __launch_bounds__(256) void gemm_proj(
    const _Float16* __restrict__ A,    // ctx16 [8192,1024]
    const _Float16* __restrict__ W,    // [1024,1024]
    const float*    __restrict__ bias, // [1024]
    float* __restrict__ out) {
  const int mbase = blockIdx.x * 128;
  const int nbase = blockIdx.y * 128;
  GEMM_CORE(A, W, acc)

#pragma unroll
  for (int nt = 0; nt < 4; ++nt) {
    const int n = nbase + ch * 64 + nt * 16 + l16;
    const float bv = bias[n];
#pragma unroll
    for (int r = 0; r < 2; ++r)
#pragma unroll
      for (int j = 0; j < 8; ++j) {
        const int m = mbase + rt * 32 + r * 16 + j + lhalf * 8;
        out[(size_t)m * Dmod + n] = acc[r][nt][j] + bv;
      }
  }
}

// ---------------------------------------------------------------------------
// Stage 2: flash attention (causal).  Grid = (S/128, B*H), 8 waves/block,
// each wave owns 16 query rows; 32-key blocks; K & V^T staged in LDS.
// ---------------------------------------------------------------------------
__global__ __launch_bounds__(256) void attn_flash(
    const _Float16* __restrict__ Qb, const _Float16* __restrict__ Kb,
    const _Float16* __restrict__ Vb, _Float16* __restrict__ ctx16) {
  __shared__ _Float16 Kl[32][72];        // 32 keys x 64 d (+8 pad)
  __shared__ _Float16 Vt[64][40];        // 64 d x 32 keys (+8 pad), transposed
  __shared__ _Float16 Pl[8][16][40];     // per-wave P (16x32 +8 pad)

  const int bh    = blockIdx.y;          // 0..63
  const int b     = bh >> 4;
  const int h     = bh & 15;
  const int qbase = blockIdx.x * 128;
  const int tid   = threadIdx.x;
  const int w     = tid >> 5;
  const int lane  = tid & 31;
  const int l16   = lane & 15;
  const int lhalf = (lane < 16) ? 0 : 1;
  const int qr0   = qbase + w * 16;

  const size_t headoff = (size_t)bh * Sseq * HDim;
  const _Float16* Qh = Qb + headoff;
  const _Float16* Kh = Kb + headoff;
  const _Float16* Vh = Vb + headoff;

  // Q A-fragments: 16x64 split into two K=32 chunks
  V16 qa[2];
  {
    const _Float16* qrow = Qh + (size_t)(qr0 + l16) * HDim;
#pragma unroll
    for (int c = 0; c < 2; ++c) {
      qa[c].h[0] = *(const h8*)(qrow + c * 32 + lhalf * 8);
      qa[c].h[1] = *(const h8*)(qrow + c * 32 + lhalf * 8 + 16);
    }
  }

  v8f ctx[4] = {};
  float mrow[8], lrow[8];
#pragma unroll
  for (int j = 0; j < 8; ++j) { mrow[j] = -1e30f; lrow[j] = 0.0f; }

  const int lkey = tid >> 3;   // 0..31
  const int ldg  = tid & 7;    // 0..7
  const int kend = qbase + 128;

  for (int kb = 0; kb < kend; kb += 32) {
    __syncthreads();
    {  // stage K (row-major) and V (transposed) tiles
      const h8 kv = *(const h8*)(Kh + (size_t)(kb + lkey) * HDim + ldg * 8);
      *(h8*)&Kl[lkey][ldg * 8] = kv;
      const h8 vv = *(const h8*)(Vh + (size_t)(kb + lkey) * HDim + ldg * 8);
#pragma unroll
      for (int i = 0; i < 8; ++i) Vt[ldg * 8 + i][lkey] = vv[i];
      if (kb + 32 < kend) {   // near-scope prefetch of next tiles
        __builtin_prefetch(Kh + (size_t)(kb + 32 + lkey) * HDim + ldg * 8, 0, 3);
        __builtin_prefetch(Vh + (size_t)(kb + 32 + lkey) * HDim + ldg * 8, 0, 3);
      }
    }
    __syncthreads();

    if (kb <= qr0 + 15) {   // wave-uniform: skip fully-masked key blocks
      // ---- scores: load all 4 K fragments, then 4 back-to-back WMMAs ----
      V16 kf[2][2];
#pragma unroll
      for (int t = 0; t < 2; ++t) {
        const _Float16* krow = &Kl[t * 16 + l16][0];
#pragma unroll
        for (int c = 0; c < 2; ++c) {
          kf[t][c].h[0] = *(const h8*)(krow + c * 32 + lhalf * 8);
          kf[t][c].h[1] = *(const h8*)(krow + c * 32 + lhalf * 8 + 16);
        }
      }
      v8f st[2] = {};
#pragma unroll
      for (int c = 0; c < 2; ++c)
#pragma unroll
        for (int t = 0; t < 2; ++t)
          st[t] = wmma_f16(qa[c].v, kf[t][c].v, st[t]);

      // ---- online softmax across the 32 new keys ----
      float p0[8], p1[8], alpha[8];
#pragma unroll
      for (int j = 0; j < 8; ++j) {
        const int row = qr0 + j + lhalf * 8;
        const bool msk0 = (kb + l16)      > row;
        const bool msk1 = (kb + 16 + l16) > row;
        const float s0 = st[0][j] * SCALE;
        const float s1 = st[1][j] * SCALE;
        float mx = fmaxf(msk0 ? -1e30f : s0, msk1 ? -1e30f : s1);
#pragma unroll
        for (int d = 1; d < 16; d <<= 1) mx = fmaxf(mx, __shfl_xor(mx, d, 32));
        const float mnew = fmaxf(mrow[j], mx);
        const float a = __expf(mrow[j] - mnew);
        mrow[j] = mnew;
        alpha[j] = a;
        p0[j] = msk0 ? 0.0f : __expf(s0 - mnew);
        p1[j] = msk1 ? 0.0f : __expf(s1 - mnew);
        float rs = p0[j] + p1[j];
#pragma unroll
        for (int d = 1; d < 16; d <<= 1) rs += __shfl_xor(rs, d, 32);
        lrow[j] = lrow[j] * a + rs;
      }
#pragma unroll
      for (int nt = 0; nt < 4; ++nt)
#pragma unroll
        for (int j = 0; j < 8; ++j) ctx[nt][j] *= alpha[j];

      // ---- V fragments first (independent), then P LDS round-trip ----
      V16 vf[4];
#pragma unroll
      for (int nt = 0; nt < 4; ++nt) {
        const _Float16* vrow = &Vt[nt * 16 + l16][0];
        vf[nt].h[0] = *(const h8*)(vrow + lhalf * 8);
        vf[nt].h[1] = *(const h8*)(vrow + lhalf * 8 + 16);
      }
#pragma unroll
      for (int j = 0; j < 8; ++j) {
        Pl[w][j + lhalf * 8][l16]      = (_Float16)p0[j];
        Pl[w][j + lhalf * 8][16 + l16] = (_Float16)p1[j];
      }
      asm volatile("s_wait_dscnt 0" ::: "memory");
      V16 pa;
      {
        const _Float16* pr = &Pl[w][l16][0];
        pa.h[0] = *(const h8*)(pr + lhalf * 8);
        pa.h[1] = *(const h8*)(pr + lhalf * 8 + 16);
      }

      // ---- ctx += P(16x32) . V(32x64), 4 back-to-back WMMAs ----
#pragma unroll
      for (int nt = 0; nt < 4; ++nt)
        ctx[nt] = wmma_f16(pa.v, vf[nt].v, ctx[nt]);
    }
  }

  // normalize + store ctx as f16 [B,S,D] (row per token, heads concatenated)
#pragma unroll
  for (int j = 0; j < 8; ++j) {
    const int row = qr0 + j + lhalf * 8;
    const float inv = 1.0f / lrow[j];
#pragma unroll
    for (int nt = 0; nt < 4; ++nt) {
      const int d = nt * 16 + l16;
      ctx16[(size_t)(b * Sseq + row) * Dmod + h * HDim + d] =
          (_Float16)(ctx[nt][j] * inv);
    }
  }
}

// ---------------------------------------------------------------------------
// Host-side launch
// ---------------------------------------------------------------------------
extern "C" void kernel_launch(void* const* d_in, const int* in_sizes, int n_in,
                              void* d_out, int out_size, void* d_ws,
                              size_t ws_size, hipStream_t stream) {
  const float* x     = (const float*)d_in[0];   // [4,2048,1024]
  const float* Wqkv  = (const float*)d_in[1];   // [3072,1024]
  const float* bqkv  = (const float*)d_in[2];   // [3072]
  const float* Wproj = (const float*)d_in[3];   // [1024,1024]
  const float* bproj = (const float*)d_in[4];   // [1024]
  // d_in[5] = causal mask (recomputed analytically on device)
  float* out = (float*)d_out;

  char* ws = (char*)d_ws;
  size_t off = 0;
  auto take = [&](size_t bytes) {
    char* p = ws + off;
    off += (bytes + 255) & ~(size_t)255;
    return p;
  };
  _Float16* x16  = (_Float16*)take((size_t)Mrows * Dmod * 2);       // 16 MB
  _Float16* Wq16 = (_Float16*)take((size_t)3 * Dmod * Dmod * 2);    //  6 MB
  _Float16* Wp16 = (_Float16*)take((size_t)Dmod * Dmod * 2);        //  2 MB
  _Float16* Qb   = (_Float16*)take((size_t)Mrows * Dmod * 2);       // 16 MB
  _Float16* Kb   = (_Float16*)take((size_t)Mrows * Dmod * 2);       // 16 MB
  _Float16* Vb   = (_Float16*)take((size_t)Mrows * Dmod * 2);       // 16 MB
  _Float16* ctx  = (_Float16*)take((size_t)Mrows * Dmod * 2);       // 16 MB

  // Stage 0: converts
  {
    int n4 = Mrows * Dmod / 4;
    cvt_f32_f16_v4<<<(n4 + 255) / 256, 256, 0, stream>>>((const f4*)x, (h4*)x16, n4);
    n4 = 3 * Dmod * Dmod / 4;
    cvt_f32_f16_v4<<<(n4 + 255) / 256, 256, 0, stream>>>((const f4*)Wqkv, (h4*)Wq16, n4);
    n4 = Dmod * Dmod / 4;
    cvt_f32_f16_v4<<<(n4 + 255) / 256, 256, 0, stream>>>((const f4*)Wproj, (h4*)Wp16, n4);
  }

  // Stage 1: QKV GEMM (grid: M/128 x N/128 = 64 x 24)
  gemm_qkv<<<dim3(Mrows / 128, (3 * Dmod) / 128), 256, 0, stream>>>(
      x16, Wq16, bqkv, Qb, Kb, Vb);

  // Stage 2: flash attention (grid: S/128 x B*H = 16 x 64)
  attn_flash<<<dim3(Sseq / 128, Bsz * Hn), 256, 0, stream>>>(Qb, Kb, Vb, ctx);

  // Stage 3: projection GEMM (grid: 64 x 8)
  gemm_proj<<<dim3(Mrows / 128, Dmod / 128), 256, 0, stream>>>(ctx, Wp16, bproj, out);
}
